// CMSSelfModifyingBlock_84164179132974
// MI455X (gfx1250) — matrix operand order repaired and checked by reading, not compiled, not run on hardware
//
#include <hip/hip_runtime.h>

typedef unsigned int uint;
typedef unsigned short u16;

#define B_ 32
#define C_ 64
#define HW_ 12544          // 112*112
#define N_ 401408          // B*HW
#define HID_ 128
#define CG_ 16             // channels per group
#define NGRP_ 8
#define KSLICE_ 2048

typedef __attribute__((ext_vector_type(16))) __bf16 v16bf;
typedef __attribute__((ext_vector_type(8)))  float  v8f;
typedef uint v4u __attribute__((ext_vector_type(4)));
typedef int  v4i __attribute__((ext_vector_type(4)));
typedef int  v8i __attribute__((ext_vector_type(8)));

#if defined(__has_builtin)
#  if __has_builtin(__builtin_amdgcn_tensor_load_to_lds)
#    define HAVE_TDM 1
#  endif
#endif

union V16U { uint4 q[2]; v16bf v; };

__device__ inline u16 f2bf(float f) {
  union { float f; uint u; } v; v.f = f;
  uint x = v.u;
  uint r = x + 0x7fffu + ((x >> 16) & 1u);   // round-to-nearest-even
  return (u16)(r >> 16);
}

__device__ inline float gelu_f(float v) {
  return 0.5f * v * (1.0f + erff(v * 0.70710678118654752f));
}
__device__ inline float gelu_grad_f(float v) {
  float cdf = 0.5f * (1.0f + erff(v * 0.70710678118654752f));
  float pdf = 0.3989422804014327f * expf(-0.5f * v * v);
  return cdf + v * pdf;
}

// A-matrix fragment (16x32 bf16): lane-hi 0 -> k {0..7,16..23}, hi 1 -> k {8..15,24..31}
__device__ inline v16bf load_a_frag(const u16* row_kc, int hi) {
  V16U u;
  const uint4* p = (const uint4*)(row_kc + hi * 8);
  u.q[0] = p[0];   // k  .. +7
  u.q[1] = p[2];   // k+16 .. +23
  return u.v;
}
// B-matrix fragment (32x16 bf16): lane-hi 0 -> k {0..7,16..23}, hi 1 -> k {8..15,24..31}
__device__ inline v16bf load_b_frag(const u16* col_kc, int hi) {
  V16U u;
  const uint4* p = (const uint4*)(col_kc + hi * 8);
  u.q[0] = p[0];
  u.q[1] = p[2];
  return u.v;
}
__device__ inline v8f wmma_bf16(v16bf a, v16bf b, v8f c) {
  return __builtin_amdgcn_wmma_f32_16x16x32_bf16(false, a, false, b, (short)0, c, false, false);
}
__device__ inline v8f v8f_zero() {
  v8f z;
#pragma unroll
  for (int i = 0; i < 8; ++i) z[i] = 0.0f;
  return z;
}

// ---------------------------------------------------------------- TDM: DMA [rows x cols] bf16
// weight matrix (row-major, stride==cols) from global into LDS. Call from ONE wave only;
// performs its own s_wait_tensorcnt. Caller must __syncthreads() afterwards.
__device__ inline void tdm_load_w_to_lds(const u16* gsrc, u16* lds_dst, int rows, int cols) {
#ifdef HAVE_TDM
  unsigned long long ga = (unsigned long long)gsrc;
  uint la = (uint)(size_t)lds_dst;   // low 32 bits of flat LDS address = LDS byte offset
  v4u g0; v8i g1; v4i g2; v4i g3;
  g0[0] = 1u;                                               // count=1, user mode
  g0[1] = la;                                               // lds_addr
  g0[2] = (uint)(ga & 0xffffffffu);                         // global_addr[31:0]
  g0[3] = (uint)((ga >> 32) & 0x01ffffffu) | 0x80000000u;   // global_addr[56:32] | type=2
  g1[0] = (1 << 16);                                        // data_size=2B (code 1)
  g1[1] = (int)((uint)cols << 16);                          // tensor_dim0[15:0] @ bits[63:48]
  g1[2] = (int)((uint)rows << 16);                          // tensor_dim1[15:0] @ bits[95:80]
  g1[3] = (int)((uint)cols << 16);                          // tile_dim0 @ bits[127:112]
  g1[4] = rows;                                             // tile_dim1 @ bits[143:128]
  g1[5] = cols;                                             // tensor_dim0_stride[31:0]
  g1[6] = 0;
  g1[7] = 0;
  g2[0] = 0; g2[1] = 0; g2[2] = 0; g2[3] = 0;
  g3[0] = 0; g3[1] = 0; g3[2] = 0; g3[3] = 0;
#  if __clang_major__ >= 23
  v8i g4; g4[0]=0; g4[1]=0; g4[2]=0; g4[3]=0; g4[4]=0; g4[5]=0; g4[6]=0; g4[7]=0;
  __builtin_amdgcn_tensor_load_to_lds(g0, g1, g2, g3, g4, 0);
#  else
  __builtin_amdgcn_tensor_load_to_lds(g0, g1, g2, g3, 0);
#  endif
  __builtin_amdgcn_s_wait_tensorcnt(0);
#else
  int lane = threadIdx.x & 31;
  int nq = (rows * cols) >> 3;    // uint4 chunks of 8 bf16
  const uint4* s = (const uint4*)gsrc;
  uint4* d = (uint4*)lds_dst;
  for (int i = lane; i < nq; i += 32) d[i] = s[i];
#endif
}

// ---------------------------------------------------------------- utilities
__global__ void k_zero(float* p, int n) {
  int i = blockIdx.x * 256 + threadIdx.x;
  if (i < n) p[i] = 0.0f;
}
__global__ void k_sgd(float* p, float* m, const float* g, int n) {
  int i = blockIdx.x * 256 + threadIdx.x;
  if (i < n) {
    float mm = 0.9f * m[i] + g[i];
    m[i] = mm;
    p[i] = p[i] - 0.01f * mm;
  }
}

// x [B,C,H,W] f32 -> Xbf [N,64] bf16 (pixel-major)
__global__ void k_convert_x(const float* x, u16* Xbf) {
  int idx = blockIdx.x * 256 + threadIdx.x;
  if (idx >= N_ * 64) return;
  int c = idx & 63;
  int n = idx >> 6;
  int b = n / HW_;
  int p = n - b * HW_;
  Xbf[idx] = f2bf(x[((size_t)b * 64 + c) * HW_ + p]);
}

// weights f32 -> bf16 (+ transposed copy of w2)
__global__ void k_wbf16(const float* w1f, const float* w2f, u16* w1bf, u16* w2bf, u16* w2tbf) {
  int i = blockIdx.x * 256 + threadIdx.x;
  if (i >= 8192) return;
  w1bf[i] = f2bf(w1f[i]);                 // [128][64]
  float w2v = w2f[i];                     // [64][128]
  w2bf[i] = f2bf(w2v);
  int c = i >> 7, o = i & 127;
  w2tbf[o * 64 + c] = f2bf(w2v);          // [128][64]
}

// ---------------------------------------------------------------- gate path
__global__ void k_pool(const float* x, float* pooled) {
  int bc = blockIdx.x;  // b*64+c
  __shared__ float sh[256];
  float s = 0.0f;
  const float* p = x + (size_t)bc * HW_;
  for (int i = threadIdx.x; i < HW_; i += 256) s += p[i];
  sh[threadIdx.x] = s;
  __syncthreads();
  for (int st = 128; st > 0; st >>= 1) {
    if (threadIdx.x < st) sh[threadIdx.x] += sh[threadIdx.x + st];
    __syncthreads();
  }
  if (threadIdx.x == 0) pooled[bc] = sh[0] / (float)HW_;
}

__global__ void k_gate(const float* pooled, const float* g1w, const float* g1b,
                       const float* g2w, const float* g2b, float* gate) {
  int b = threadIdx.x;
  if (b >= 32) return;
  float acc = 0.0f;
  for (int i = 0; i < 16; ++i) {
    float s = g1b[i];
    for (int c = 0; c < 64; ++c) s += pooled[b * 64 + c] * g1w[i * 64 + c];
    acc += gelu_f(s) * g2w[i];
  }
  acc += g2b[0];
  gate[b] = 1.0f / (1.0f + expf(-acc));
}

// ---------------------------------------------------------------- conv1: H1[N,128] = Xbf @ w1bf^T
__global__ __launch_bounds__(128) void k_conv1(const u16* X, const u16* W1bf, float* H1) {
  __shared__ u16 lw[8192];
  if ((threadIdx.x >> 5) == 0) tdm_load_w_to_lds(W1bf, lw, 128, 64);
  __syncthreads();

  int wid = threadIdx.x >> 5, lane = threadIdx.x & 31;
  int lo = lane & 15, hi = lane >> 4;
  int row0 = blockIdx.x * 64 + wid * 16;
  v8f acc[8];
#pragma unroll
  for (int t = 0; t < 8; ++t) acc[t] = v8f_zero();
  const u16* arow = X + (size_t)(row0 + lo) * 64;
#pragma unroll
  for (int kc = 0; kc < 2; ++kc) {
    v16bf a = load_a_frag(arow + kc * 32, hi);
    v16bf bf[8];
#pragma unroll
    for (int ot = 0; ot < 8; ++ot)
      bf[ot] = load_b_frag(lw + (ot * 16 + lo) * 64 + kc * 32, hi);
#pragma unroll
    for (int ot = 0; ot < 8; ++ot)
      acc[ot] = wmma_bf16(a, bf[ot], acc[ot]);
  }
#pragma unroll
  for (int ot = 0; ot < 8; ++ot) {
    int col = ot * 16 + lo;
#pragma unroll
    for (int r = 0; r < 8; ++r) {
      int m = r + hi * 8;
      H1[(size_t)(row0 + m) * 128 + col] = acc[ot][r];
    }
  }
}

// ---------------------------------------------------------------- GroupNorm stats (biased var)
__global__ void k_gn_stats(const float* H1, float* mu, float* rsig) {
  int bg = blockIdx.x;
  int b = bg >> 3, g = bg & 7;
  __shared__ float ssum[256], ssq[256];
  float s = 0.0f, sq = 0.0f;
  size_t base = (size_t)b * HW_ * 128 + g * 16;
  for (int p = threadIdx.x; p < HW_; p += 256) {
    const float* ptr = H1 + base + (size_t)p * 128;
#pragma unroll
    for (int j = 0; j < 16; ++j) { float v = ptr[j]; s += v; sq += v * v; }
  }
  ssum[threadIdx.x] = s; ssq[threadIdx.x] = sq;
  __syncthreads();
  for (int st = 128; st > 0; st >>= 1) {
    if (threadIdx.x < st) { ssum[threadIdx.x] += ssum[threadIdx.x + st]; ssq[threadIdx.x] += ssq[threadIdx.x + st]; }
    __syncthreads();
  }
  if (threadIdx.x == 0) {
    float M = (float)(CG_ * HW_);
    float m = ssum[0] / M;
    float var = ssq[0] / M - m * m;
    mu[bg] = m;
    rsig[bg] = rsqrtf(var + 1e-5f);
  }
}

// ---------------------------------------------------------------- GN affine + GELU -> Abf bf16
__global__ void k_act(const float* H1, const float* mu, const float* rsig,
                      const float* gnw, const float* gnb, u16* Abf) {
  int idx = blockIdx.x * 256 + threadIdx.x;
  if (idx >= N_ * 128) return;
  int ch = idx & 127;
  int n = idx >> 7;
  int b = n / HW_;
  int bg = b * 8 + (ch >> 4);
  float v = (H1[idx] - mu[bg]) * rsig[bg] * gnw[ch] + gnb[ch];
  Abf[idx] = f2bf(gelu_f(v));
}

// ---------------------------------------------------------------- conv2: Y[N,64] = Abf @ w2bf^T (+x)
__global__ __launch_bounds__(128) void k_conv2(const u16* Abf, const u16* W2bf,
                                               const float* x, float* Y, int addres) {
  __shared__ u16 lw[8192];
  if ((threadIdx.x >> 5) == 0) tdm_load_w_to_lds(W2bf, lw, 64, 128);
  __syncthreads();

  int wid = threadIdx.x >> 5, lane = threadIdx.x & 31;
  int lo = lane & 15, hi = lane >> 4;
  int row0 = blockIdx.x * 64 + wid * 16;
  v8f acc[4];
#pragma unroll
  for (int t = 0; t < 4; ++t) acc[t] = v8f_zero();
  const u16* arow = Abf + (size_t)(row0 + lo) * 128;
#pragma unroll
  for (int kc = 0; kc < 4; ++kc) {
    v16bf a = load_a_frag(arow + kc * 32, hi);
    v16bf bf[4];
#pragma unroll
    for (int ot = 0; ot < 4; ++ot)
      bf[ot] = load_b_frag(lw + (ot * 16 + lo) * 128 + kc * 32, hi);
#pragma unroll
    for (int ot = 0; ot < 4; ++ot)
      acc[ot] = wmma_bf16(a, bf[ot], acc[ot]);
  }
#pragma unroll
  for (int ot = 0; ot < 4; ++ot) {
    int col = ot * 16 + lo;
#pragma unroll
    for (int r = 0; r < 8; ++r) {
      int row = row0 + r + hi * 8;
      float v = acc[ot][r];
      if (addres) {
        int bb = row / HW_;
        int p = row - bb * HW_;
        v += x[((size_t)bb * 64 + col) * HW_ + p];
      }
      Y[(size_t)row * 64 + col] = v;
    }
  }
}

// ---------------------------------------------------------------- channel stats -> surprise-grad coeffs
__global__ void k_ch_stats(const float* Y, const float* rm, const float* rv,
                           float* cm, float* ac, float* bc) {
  int c = blockIdx.x;
  __shared__ float ssum[256], ssq[256];
  float s = 0.0f, sq = 0.0f;
  for (int n = threadIdx.x; n < N_; n += 256) {
    float v = Y[(size_t)n * 64 + c];
    s += v; sq += v * v;
  }
  ssum[threadIdx.x] = s; ssq[threadIdx.x] = sq;
  __syncthreads();
  for (int st = 128; st > 0; st >>= 1) {
    if (threadIdx.x < st) { ssum[threadIdx.x] += ssum[threadIdx.x + st]; ssq[threadIdx.x] += ssq[threadIdx.x + st]; }
    __syncthreads();
  }
  if (threadIdx.x == 0) {
    float mean = ssum[0] / (float)N_;
    float var = (ssq[0] - (float)N_ * mean * mean) / ((float)N_ - 1.0f);  // unbiased
    cm[c] = mean;
    float dLdcm = 2.0f * (mean - rm[c]) / 64.0f;
    float denom = rv[c] + 1e-8f;
    float dLdcv = 2.0f * (var / denom - 1.0f) / (64.0f * denom);
    ac[c] = dLdcm / (float)N_;
    bc[c] = dLdcv * 2.0f / ((float)N_ - 1.0f);
  }
}

__global__ void k_dY(const float* Y, const float* cm, const float* ac, const float* bc, u16* dYbf) {
  int idx = blockIdx.x * 256 + threadIdx.x;
  if (idx >= N_ * 64) return;
  int c = idx & 63;
  dYbf[idx] = f2bf(ac[c] + bc[c] * (Y[idx] - cm[c]));
}

// ---------------------------------------------------------------- split-K A^T B  (weight grads)
template <int MA, int NB>
__global__ __launch_bounds__(128) void k_gemm_atb(const u16* Abuf, const u16* Bbuf, float* Gout) {
  __shared__ u16 lat[MA * 32];
  __shared__ u16 lbt[NB * 32];
  int tid = threadIdx.x;
  int wid = tid >> 5, lane = tid & 31, lo = lane & 15, hi = lane >> 4;
  v8f acc[8];
#pragma unroll
  for (int t = 0; t < 8; ++t) acc[t] = v8f_zero();
  int n0 = blockIdx.x * KSLICE_;
  for (int ck = 0; ck < KSLICE_; ck += 32) {
    int nb = n0 + ck;
    if (nb + 64 < N_) __builtin_prefetch(Abuf + (size_t)(nb + 32) * MA, 0, 1);
    for (int i = tid; i < MA * 32; i += 128) {
      int nn = i / MA, m = i - nn * MA;
      lat[m * 32 + nn] = Abuf[(size_t)(nb + nn) * MA + m];
    }
    for (int i = tid; i < NB * 32; i += 128) {
      int nn = i / NB, m = i - nn * NB;
      lbt[m * 32 + nn] = Bbuf[(size_t)(nb + nn) * NB + m];
    }
    __syncthreads();
#pragma unroll
    for (int t = 0; t < 8; ++t) {
      int tileid = wid * 8 + t;
      int mt = tileid / (NB / 16), nt = tileid - mt * (NB / 16);
      v16bf a = load_a_frag(&lat[(mt * 16 + lo) * 32], hi);
      v16bf b = load_b_frag(&lbt[(nt * 16 + lo) * 32], hi);
      acc[t] = wmma_bf16(a, b, acc[t]);
    }
    __syncthreads();
  }
#pragma unroll
  for (int t = 0; t < 8; ++t) {
    int tileid = wid * 8 + t;
    int mt = tileid / (NB / 16), nt = tileid - mt * (NB / 16);
    int col = nt * 16 + lo;
#pragma unroll
    for (int r = 0; r < 8; ++r) {
      int m = mt * 16 + r + hi * 8;
      atomicAdd(&Gout[m * NB + col], acc[t][r]);
    }
  }
}

// ---------------------------------------------------------------- dA = dY @ w2 then * gelu'(haff)
__global__ __launch_bounds__(128) void k_dA(const u16* dYbf, const u16* W2Tbf,
                                            const float* H1, const float* mu, const float* rsig,
                                            const float* gnw, const float* gnb, float* dHaff) {
  __shared__ u16 lw[8192];
  if ((threadIdx.x >> 5) == 0) tdm_load_w_to_lds(W2Tbf, lw, 128, 64);
  __syncthreads();

  int wid = threadIdx.x >> 5, lane = threadIdx.x & 31;
  int lo = lane & 15, hi = lane >> 4;
  int row0 = blockIdx.x * 64 + wid * 16;
  v8f acc[8];
#pragma unroll
  for (int t = 0; t < 8; ++t) acc[t] = v8f_zero();
  const u16* arow = dYbf + (size_t)(row0 + lo) * 64;
#pragma unroll
  for (int kc = 0; kc < 2; ++kc) {
    v16bf a = load_a_frag(arow + kc * 32, hi);
    v16bf bf[8];
#pragma unroll
    for (int ot = 0; ot < 8; ++ot)
      bf[ot] = load_b_frag(lw + (ot * 16 + lo) * 64 + kc * 32, hi);
#pragma unroll
    for (int ot = 0; ot < 8; ++ot)
      acc[ot] = wmma_bf16(a, bf[ot], acc[ot]);
  }
#pragma unroll
  for (int ot = 0; ot < 8; ++ot) {
    int ch = ot * 16 + lo;
#pragma unroll
    for (int r = 0; r < 8; ++r) {
      int row = row0 + r + hi * 8;
      int b = row / HW_;
      int bg = b * 8 + (ch >> 4);
      float hv = (H1[(size_t)row * 128 + ch] - mu[bg]) * rsig[bg] * gnw[ch] + gnb[ch];
      dHaff[(size_t)row * 128 + ch] = acc[ot][r] * gelu_grad_f(hv);
    }
  }
}

// ---------------------------------------------------------------- GN backward sums + affine grads
__global__ void k_gnback(const float* dHaff, const float* H1, const float* mu, const float* rsig,
                         const float* gnw, float* S1, float* S2, float* gGnw, float* gGnb) {
  int bg = blockIdx.x;
  int b = bg >> 3, g = bg & 7;
  int j = threadIdx.x & 15, pl = threadIdx.x >> 4;
  int ch = g * 16 + j;
  float w = gnw[ch], m = mu[bg], rs = rsig[bg];
  float s1 = 0.0f, s2 = 0.0f, ga = 0.0f, gb = 0.0f;
  size_t base = (size_t)b * HW_ * 128 + ch;
  for (int p = pl; p < HW_; p += 16) {
    float h = H1[base + (size_t)p * 128];
    float d = dHaff[base + (size_t)p * 128];
    float s = (h - m) * rs;
    float ds = d * w;
    s1 += ds; s2 += ds * s; ga += d * s; gb += d;
  }
  __shared__ float sh[512];
  sh[threadIdx.x] = s1; sh[256 + threadIdx.x] = s2;
  __syncthreads();
  for (int st = 128; st > 0; st >>= 1) {
    if (threadIdx.x < st) { sh[threadIdx.x] += sh[threadIdx.x + st]; sh[256 + threadIdx.x] += sh[256 + threadIdx.x + st]; }
    __syncthreads();
  }
  if (threadIdx.x == 0) { S1[bg] = sh[0]; S2[bg] = sh[256]; }
  __syncthreads();
  sh[threadIdx.x] = ga; sh[256 + threadIdx.x] = gb;
  __syncthreads();
  for (int st = 8; st > 0; st >>= 1) {
    if (pl < st) {
      sh[pl * 16 + j] += sh[(pl + st) * 16 + j];
      sh[256 + pl * 16 + j] += sh[256 + (pl + st) * 16 + j];
    }
    __syncthreads();
  }
  if (pl == 0) {
    atomicAdd(&gGnw[ch], sh[j]);
    atomicAdd(&gGnb[ch], sh[256 + j]);
  }
}

// ---------------------------------------------------------------- dH1 (GN input grad) -> bf16
__global__ void k_dH1(const float* dHaff, const float* H1, const float* mu, const float* rsig,
                      const float* gnw, const float* S1, const float* S2, u16* dH1bf) {
  int idx = blockIdx.x * 256 + threadIdx.x;
  if (idx >= N_ * 128) return;
  int ch = idx & 127;
  int n = idx >> 7;
  int b = n / HW_;
  int bg = b * 8 + (ch >> 4);
  const float invM = 1.0f / (float)(CG_ * HW_);
  float s = (H1[idx] - mu[bg]) * rsig[bg];
  float ds = dHaff[idx] * gnw[ch];
  float dh1 = rsig[bg] * (ds - S1[bg] * invM - s * (S2[bg] * invM));
  dH1bf[idx] = f2bf(dh1);
}

// ---------------------------------------------------------------- final combine
__global__ void k_out(const float* x, const float* Mod, const float* gate, const float* rs, float* out) {
  int idx = blockIdx.x * 256 + threadIdx.x;
  if (idx >= B_ * C_ * HW_) return;
  int p = idx % HW_;
  int bcc = idx / HW_;
  int c = bcc & 63;
  int b = bcc >> 6;
  int n = b * HW_ + p;
  out[idx] = x[idx] + rs[0] * gate[b] * Mod[(size_t)n * 64 + c];
}

// ================================================================ host
static inline size_t align256(size_t v) { return (v + 255) & ~(size_t)255; }

extern "C" void kernel_launch(void* const* d_in, const int* in_sizes, int n_in,
                              void* d_out, int out_size, void* d_ws, size_t ws_size,
                              hipStream_t stream) {
  const float* x    = (const float*)d_in[0];
  const float* w1   = (const float*)d_in[1];
  const float* gnw0 = (const float*)d_in[2];
  const float* gnb0 = (const float*)d_in[3];
  const float* w2   = (const float*)d_in[4];
  const float* g1w  = (const float*)d_in[5];
  const float* g1b  = (const float*)d_in[6];
  const float* g2w  = (const float*)d_in[7];
  const float* g2b  = (const float*)d_in[8];
  const float* rm   = (const float*)d_in[9];
  const float* rv   = (const float*)d_in[10];
  const float* rs   = (const float*)d_in[11];
  float* out = (float*)d_out;

  char* ws = (char*)d_ws;
  size_t off = 0;
  auto alloc = [&](size_t bytes) -> void* {
    void* p = ws + off;
    off = align256(off + bytes);
    return p;
  };

  u16*   Xbf    = (u16*)  alloc((size_t)N_ * 64 * 2);
  float* H1     = (float*)alloc((size_t)N_ * 128 * 4);
  float* DHAFF  = (float*)alloc((size_t)N_ * 128 * 4);
  u16*   ABUF   = (u16*)  alloc((size_t)N_ * 128 * 2);   // Abf, later dH1bf
  float* YBUF   = (float*)alloc((size_t)N_ * 64 * 4);
  u16*   DYBF   = (u16*)  alloc((size_t)N_ * 64 * 2);
  float* params = (float*)alloc(16640 * 4);              // w1(8192) gnw(128) gnb(128) w2(8192)
  float* moms   = (float*)alloc(16640 * 4);
  float* grads  = (float*)alloc(16640 * 4);
  u16*   w1bf   = (u16*)  alloc(8192 * 2);
  u16*   w2bf   = (u16*)  alloc(8192 * 2);
  u16*   w2tbf  = (u16*)  alloc(8192 * 2);
  float* mu     = (float*)alloc(256 * 4);
  float* rsig   = (float*)alloc(256 * 4);
  float* S1     = (float*)alloc(256 * 4);
  float* S2     = (float*)alloc(256 * 4);
  float* cm     = (float*)alloc(64 * 4);
  float* ac     = (float*)alloc(64 * 4);
  float* bc     = (float*)alloc(64 * 4);
  float* pooled = (float*)alloc(2048 * 4);
  float* gate   = (float*)alloc(32 * 4);

  float* w1f  = params;
  float* gnwf = params + 8192;
  float* gnbf = params + 8320;
  float* w2f  = params + 8448;
  float* gW1  = grads;
  float* gGnw = grads + 8192;
  float* gGnb = grads + 8320;
  float* gW2  = grads + 8448;

  // --- gate path + input conversion
  k_convert_x<<<(N_ * 64) / 256, 256, 0, stream>>>(x, Xbf);
  k_pool<<<B_ * C_, 256, 0, stream>>>(x, pooled);
  k_gate<<<1, 32, 0, stream>>>(pooled, g1w, g1b, g2w, g2b, gate);

  // --- init params / momenta
  hipMemcpyAsync(w1f,  w1,   8192 * 4, hipMemcpyDeviceToDevice, stream);
  hipMemcpyAsync(gnwf, gnw0, 128 * 4,  hipMemcpyDeviceToDevice, stream);
  hipMemcpyAsync(gnbf, gnb0, 128 * 4,  hipMemcpyDeviceToDevice, stream);
  hipMemcpyAsync(w2f,  w2,   8192 * 4, hipMemcpyDeviceToDevice, stream);
  k_zero<<<(16640 + 255) / 256, 256, 0, stream>>>(moms, 16640);

  const int gemmBlocks = N_ / 64;       // 6272
  const int atbBlocks  = N_ / KSLICE_;  // 196

  // --- 2 inner SGD steps
  for (int step = 0; step < 2; ++step) {
    k_wbf16<<<32, 256, 0, stream>>>(w1f, w2f, w1bf, w2bf, w2tbf);
    k_conv1<<<gemmBlocks, 128, 0, stream>>>(Xbf, w1bf, H1);
    k_gn_stats<<<B_ * NGRP_, 256, 0, stream>>>(H1, mu, rsig);
    k_act<<<(N_ * 128) / 256, 256, 0, stream>>>(H1, mu, rsig, gnwf, gnbf, ABUF);
    k_conv2<<<gemmBlocks, 128, 0, stream>>>(ABUF, w2bf, x, YBUF, 1);
    k_ch_stats<<<64, 256, 0, stream>>>(YBUF, rm, rv, cm, ac, bc);
    k_dY<<<(N_ * 64) / 256, 256, 0, stream>>>(YBUF, cm, ac, bc, DYBF);
    k_zero<<<(16640 + 255) / 256, 256, 0, stream>>>(grads, 16640);
    k_gemm_atb<64, 128><<<atbBlocks, 128, 0, stream>>>(DYBF, ABUF, gW2);
    k_dA<<<gemmBlocks, 128, 0, stream>>>(DYBF, w2tbf, H1, mu, rsig, gnwf, gnbf, DHAFF);
    k_gnback<<<B_ * NGRP_, 256, 0, stream>>>(DHAFF, H1, mu, rsig, gnwf, S1, S2, gGnw, gGnb);
    k_dH1<<<(N_ * 128) / 256, 256, 0, stream>>>(DHAFF, H1, mu, rsig, gnwf, S1, S2, ABUF);
    k_gemm_atb<128, 64><<<atbBlocks, 128, 0, stream>>>(ABUF, Xbf, gW1);
    k_sgd<<<(16640 + 255) / 256, 256, 0, stream>>>(params, moms, grads, 16640);
  }

  // --- final modifier with updated params
  k_wbf16<<<32, 256, 0, stream>>>(w1f, w2f, w1bf, w2bf, w2tbf);
  k_conv1<<<gemmBlocks, 128, 0, stream>>>(Xbf, w1bf, H1);
  k_gn_stats<<<B_ * NGRP_, 256, 0, stream>>>(H1, mu, rsig);
  k_act<<<(N_ * 128) / 256, 256, 0, stream>>>(H1, mu, rsig, gnwf, gnbf, ABUF);
  k_conv2<<<gemmBlocks, 128, 0, stream>>>(ABUF, w2bf, x, YBUF, 0);   // modification only
  k_out<<<(B_ * C_ * HW_ + 255) / 256, 256, 0, stream>>>(x, YBUF, gate, rs, out);
}
